// W8A8BFP32O32LinearWithoutScaling_32968168964169
// MI455X (gfx1250) — 8-bit matrix operand order repaired and checked by reading, compiled, not run on hardware
//
#include <hip/hip_runtime.h>


typedef __attribute__((ext_vector_type(8))) int v8i;
typedef __attribute__((ext_vector_type(4))) int v4i;
typedef __attribute__((ext_vector_type(2))) int v2i;

#define M_DIM 8192
#define N_DIM 4096
#define K_DIM 4096
#define BM    128
#define BN    256
#define KT    64
#define NK    (K_DIM / KT)
#define LDS_ROW 80
#define A_TILE_BYTES (BM * LDS_ROW)
#define B_TILE_BYTES (BN * LDS_ROW)
#define B_BASE      (2 * A_TILE_BYTES)
#define SMEM_BYTES  (2 * A_TILE_BYTES + 2 * B_TILE_BYTES)

#if defined(__has_builtin)
#  if __has_builtin(__builtin_amdgcn_global_load_async_to_lds_b128)
#    define HAVE_ASYNC_B128 1
#  endif
#  if __has_builtin(__builtin_amdgcn_s_wait_asynccnt)
#    define HAVE_WAIT_ASYNC 1
#  endif
#endif

typedef __attribute__((address_space(1))) v4i GV4;
typedef __attribute__((address_space(3))) v4i LV4;

__device__ __forceinline__ void copy_b128_to_lds(const void* g, void* l) {
#if defined(HAVE_ASYNC_B128)
  __builtin_amdgcn_global_load_async_to_lds_b128((GV4*)g, (LV4*)l, 0, 0);
#else
  *(v4i*)l = *(const v4i*)g;
#endif
}

__device__ __forceinline__ void wait_async_zero() {
#if defined(HAVE_ASYNC_B128)
#  if defined(HAVE_WAIT_ASYNC)
  __builtin_amdgcn_s_wait_asynccnt(0);
#  else
  asm volatile("s_wait_asynccnt 0" ::: "memory");
#  endif
#endif
}

__device__ __forceinline__ void issue_tile(const signed char* __restrict__ x,
                                           const signed char* __restrict__ w,
                                           unsigned char* La, unsigned char* Lb,
                                           int tid, int mb, int nb, int k0) {
#pragma unroll
  for (int i = 0; i < 2; ++i) {
    const int c   = tid + i * 256;
    const int row = c >> 2;
    const int ko  = (c & 3) * 16;
    copy_b128_to_lds(x + (size_t)(mb + row) * (size_t)K_DIM + (size_t)(k0 + ko),
                     La + row * LDS_ROW + ko);
  }
#pragma unroll
  for (int i = 0; i < 4; ++i) {
    const int c   = tid + i * 256;
    const int row = c >> 2;
    const int ko  = (c & 3) * 16;
    copy_b128_to_lds(w + (size_t)(nb + row) * (size_t)K_DIM + (size_t)(k0 + ko),
                     Lb + row * LDS_ROW + ko);
  }
}

__global__ __launch_bounds__(256)
void w8a8_o32_wmma_kernel(const signed char* __restrict__ x,
                          const signed char* __restrict__ w,
                          const float* __restrict__ bias,
                          float* __restrict__ out) {
  __shared__ __align__(16) unsigned char smem[SMEM_BYTES];

  const int tid  = threadIdx.x;
  const int lane = tid & 31;
  const int wv   = tid >> 5;
  const int wm   = wv >> 2;
  const int wn   = wv & 3;
  const int half = lane >> 4;
  const int l15  = lane & 15;

  const int mb = blockIdx.y * BM;
  const int nb = blockIdx.x * BN;

  v8i acc[4][4] = {};

  issue_tile(x, w, smem, smem + B_BASE, tid, mb, nb, 0);

  int buf = 0;
  for (int kt = 0; kt < NK; ++kt) {
    wait_async_zero();
    __syncthreads();

    if (kt + 1 < NK) {
      issue_tile(x, w,
                 smem + (buf ^ 1) * A_TILE_BYTES,
                 smem + B_BASE + (buf ^ 1) * B_TILE_BYTES,
                 tid, mb, nb, (kt + 1) * KT);
    }

    const unsigned char* As = smem + buf * A_TILE_BYTES;
    const unsigned char* Bs = smem + B_BASE + buf * B_TILE_BYTES;

    v8i af[4];
#pragma unroll
    for (int t = 0; t < 4; ++t) {
      const unsigned char* rp = As + (wm * 64 + t * 16 + l15) * LDS_ROW;
#pragma unroll
      for (int p = 0; p < 4; ++p) {
        v2i d = *(const v2i*)(rp + (p >> 1) * 32 + half * 16 + (p & 1) * 8);
        af[t][2 * p]     = d.x;
        af[t][2 * p + 1] = d.y;
      }
    }

    v8i bf[4];
#pragma unroll
    for (int u = 0; u < 4; ++u) {
      const unsigned char* rp = Bs + (wn * 64 + u * 16 + l15) * LDS_ROW;
      v4i lo = *(const v4i*)(rp + half * 16);
      v4i hi = *(const v4i*)(rp + 32 + half * 16);
      bf[u][0] = lo.x; bf[u][1] = lo.y; bf[u][2] = lo.z; bf[u][3] = lo.w;
      bf[u][4] = hi.x; bf[u][5] = hi.y; bf[u][6] = hi.z; bf[u][7] = hi.w;
    }

#pragma unroll
    for (int u = 0; u < 4; ++u)
#pragma unroll
      for (int t = 0; t < 4; ++t)
        acc[t][u] = __builtin_amdgcn_wmma_i32_16x16x64_iu8(
             true, af[t],  true, bf[u], acc[t][u],
             false,  false);

    buf ^= 1;
  }

#pragma unroll
  for (int u = 0; u < 4; ++u) {
    const int n = nb + wn * 64 + u * 16 + l15;
    const float bv = rintf(bias[n]);
#pragma unroll
    for (int t = 0; t < 4; ++t) {
      const int m0 = mb + wm * 64 + t * 16 + half * 8;
#pragma unroll
      for (int r = 0; r < 8; ++r) {
        out[(size_t)(m0 + r) * (size_t)N_DIM + n] = (float)acc[t][u][r] + bv;
      }
    }
  }
}

extern "C" void kernel_launch(void* const* d_in, const int* in_sizes, int n_in,
                              void* d_out, int out_size, void* d_ws, size_t ws_size,
                              hipStream_t stream) {
  (void)in_sizes; (void)n_in; (void)out_size; (void)d_ws; (void)ws_size;
  const signed char* x    = (const signed char*)d_in[0];
  const signed char* wgt  = (const signed char*)d_in[1];
  const float*       bias = (const float*)d_in[2];
  float*             out  = (float*)d_out;

  dim3 grid(N_DIM / BN, M_DIM / BM);
  w8a8_o32_wmma_kernel<<<grid, dim3(256), 0, stream>>>(x, wgt, bias, out);
}
